// HumanVAttention_9740985827422
// MI455X (gfx1250) — matrix operand order repaired and checked by reading, not compiled, not run on hardware
//
#include <hip/hip_runtime.h>
#include <hip/hip_bf16.h>
#include <math.h>

typedef __attribute__((ext_vector_type(8)))  float        v8f;
typedef __attribute__((ext_vector_type(16))) __bf16       v16bf;
typedef __attribute__((ext_vector_type(4)))  unsigned int u32x4;
typedef __attribute__((ext_vector_type(8)))  unsigned int u32x8;

#define B_    2
#define S_    4096
#define D_    2048
#define H_    16
#define HD_   128
#define BS_   64
#define NB_   64          // S_/BS_
#define NS_   6           // GNB(2) + LNB(4)
#define NSK_  384         // NS_*BS_
#define SROW_ 392         // LDS score row stride (floats), padded
#define SCALE_ 0.08838834764831845f   // 128^-0.5
#define NEG_  -1e9f

// attention LDS layout (bytes)
#define LDS_K_OFF 0                      // 6 * 64 * 128 * 2 = 98304
#define LDS_V_OFF 98304                  // 6 * 128 * 64 * 2 = 98304
#define LDS_S_OFF 196608                 // 64 * SROW_ * 4   = 100352
#define LDS_TOTAL (196608 + 64 * SROW_ * 4)

union FragBF { v16bf v; u32x4 u[2]; };

// ---------------------------------------------------------------------------
// TDM: issue a 2D tensor-tile load into LDS (D# per CDNA5 ISA ch.8)
//   group0: count=1, lds_addr, global_addr, type=2
//   group1: data_size=2B, tensor_dim0/1, tile_dim0/1, dim0 stride
//   groups 2/3: zero (2D tensor)
// ---------------------------------------------------------------------------
__device__ __forceinline__ void tdm_load_2d(unsigned lds_addr, const void* gptr,
                                            unsigned tdim0, unsigned tdim1,
                                            unsigned stride0,
                                            unsigned tile0, unsigned tile1) {
  unsigned long long ga = (unsigned long long)(size_t)gptr;
  u32x4 g0;
  g0[0] = 1u;                                   // count=1, user mode
  g0[1] = lds_addr;                             // LDS byte address
  g0[2] = (unsigned)ga;                         // global_addr[31:0]
  g0[3] = (unsigned)((ga >> 32) & 0x01FFFFFFu) | (2u << 30);  // addr[56:32] | type=2
  u32x8 g1;
  g1[0] = 1u << 16;                             // wg_mask=0, data_size=1 (2 bytes)
  g1[1] = (tdim0 & 0xFFFFu) << 16;              // tensor_dim0[15:0] -> bits 63:48
  g1[2] = (tdim0 >> 16) | ((tdim1 & 0xFFFFu) << 16);
  g1[3] = (tdim1 >> 16) | (tile0 << 16);        // tile_dim0 -> bits 127:112
  g1[4] = tile1;                                // tile_dim1 (tile_dim2 = 0)
  g1[5] = stride0;                              // tensor_dim0_stride[31:0]
  g1[6] = 0u;                                   // stride0 hi | stride1 lo
  g1[7] = 0u;                                   // stride1 hi
  u32x4 gz = (u32x4)0;
  asm volatile("tensor_load_to_lds %0, %1, %2, %3"
               :: "s"(g0), "s"(g1), "s"(gz), "s"(gz)
               : "memory");
}

// ---------------------------------------------------------------------------
// f32 -> bf16 conversion (same layout)
// ---------------------------------------------------------------------------
__global__ __launch_bounds__(256) void cvt_f32_bf16(const float* __restrict__ in,
                                                    __bf16* __restrict__ out, int n) {
  int i = blockIdx.x * blockDim.x + threadIdx.x;
  if (i < n) out[i] = (__bf16)in[i];
}

// ---------------------------------------------------------------------------
// W [K,N] f32  ->  WT [N,K] bf16  (pre-transpose so WMMA B-frags are contiguous)
// ---------------------------------------------------------------------------
__global__ __launch_bounds__(256) void transpose_w_bf16(const float* __restrict__ W,
                                                        __bf16* __restrict__ WT,
                                                        int K, int N) {
  int i = blockIdx.x * blockDim.x + threadIdx.x;
  if (i >= K * N) return;
  int n = i / K, k = i % K;
  WT[(size_t)n * K + k] = (__bf16)W[(size_t)k * N + n];
}

// ---------------------------------------------------------------------------
// C[M,N] f32 = A[M,K] bf16 @ BT[N,K] bf16   (wave computes a 64x64 C tile:
// 16 WMMAs per 16 b128 loads per K-step -> ~32 FLOP/byte from L2)
// ---------------------------------------------------------------------------
__global__ __launch_bounds__(256) void gemm_bf16_wmma(const __bf16* __restrict__ A,
                                                      const __bf16* __restrict__ BT,
                                                      float* __restrict__ C,
                                                      int M, int N, int K) {
  const int wave = threadIdx.x >> 5, lane = threadIdx.x & 31;
  const int lm = lane & 15, g = lane >> 4;
  const int tilesN = N >> 6;
  const int tile = blockIdx.x * 8 + wave;
  if (tile >= (M >> 6) * tilesN) return;
  const int m0 = (tile / tilesN) << 6, n0 = (tile % tilesN) << 6;

  v8f acc[4][4];
#pragma unroll
  for (int i = 0; i < 4; i++)
#pragma unroll
    for (int j = 0; j < 4; j++) acc[i][j] = (v8f)0.f;

  const __bf16* arow[4];
  const __bf16* brow[4];
#pragma unroll
  for (int i = 0; i < 4; i++) {
    arow[i] = A + (size_t)(m0 + i * 16 + lm) * K;
    brow[i] = BT + (size_t)(n0 + i * 16 + lm) * K;
  }

  for (int k0 = 0; k0 < K; k0 += 32) {
    FragBF a[4], b[4];
#pragma unroll
    for (int i = 0; i < 4; i++) {
      const u32x4* pa = (const u32x4*)(arow[i] + k0 + g * 8);
      a[i].u[0] = pa[0]; a[i].u[1] = pa[2];      // K: [g*8,+8) and [16+g*8,+8)
      const u32x4* pb = (const u32x4*)(brow[i] + k0 + g * 8);
      b[i].u[0] = pb[0]; b[i].u[1] = pb[2];      // K: [g*8,+8) and [16+g*8,+8)
    }
    __builtin_prefetch(arow[0] + k0 + 64, 0, 1); // -> global_prefetch_b8
    __builtin_prefetch(brow[0] + k0 + 64, 0, 1);
#pragma unroll
    for (int i = 0; i < 4; i++)
#pragma unroll
      for (int j = 0; j < 4; j++)
        acc[i][j] = __builtin_amdgcn_wmma_f32_16x16x32_bf16(
            false, a[i].v, false, b[j].v, (short)0, acc[i][j], false, false);
  }
  // D layout: lane = N col, VGPR v -> M = v + 8*(lane/16)
#pragma unroll
  for (int i = 0; i < 4; i++)
#pragma unroll
    for (int j = 0; j < 4; j++)
#pragma unroll
      for (int v = 0; v < 8; v++)
        C[(size_t)(m0 + i * 16 + v + 8 * g) * N + n0 + j * 16 + lm] = acc[i][j][v];
}

// ---------------------------------------------------------------------------
// phase[m,h] = hidden[m,:] . Wphase[:,h]     (tiny: N=16)
// ---------------------------------------------------------------------------
__global__ __launch_bounds__(256) void phase_proj(const float* __restrict__ hid,
                                                  const float* __restrict__ Wp,
                                                  float* __restrict__ phase) {
  int i = blockIdx.x * blockDim.x + threadIdx.x;   // [0, 8192*16)
  if (i >= (B_ * S_) * H_) return;
  int m = i / H_, h = i % H_;
  const float* a = hid + (size_t)m * D_;
  float s = 0.f;
  for (int k = 0; k < D_; k++) s += a[k] * Wp[(size_t)k * H_ + h];
  phase[i] = s;
}

// ---------------------------------------------------------------------------
// Learned-phase RoPE; writes Qr/Kr as bf16 [B,H,S,HD]
// ---------------------------------------------------------------------------
__global__ __launch_bounds__(256) void rope_kernel(const float* __restrict__ qg,
                                                   const float* __restrict__ kg,
                                                   const float* __restrict__ phase,
                                                   const float* __restrict__ cosT,
                                                   const float* __restrict__ sinT,
                                                   __bf16* __restrict__ Qr,
                                                   __bf16* __restrict__ Kr) {
  int i = blockIdx.x * blockDim.x + threadIdx.x;   // [0, B*H*S)
  if (i >= B_ * H_ * S_) return;
  int b = i / (H_ * S_);
  int h = (i / S_) % H_;
  int s = i % S_;
  size_t bs = (size_t)b * S_ + s;
  float ph = phase[bs * H_ + h];
  float delta = tanhf(ph) * 0.25f;
  float cd = cosf(delta), sd = sinf(delta);
  const float* qr = qg + bs * (H_ * HD_) + h * HD_;
  const float* kr = kg + bs * (H_ * HD_) + h * HD_;
  const float* cr = cosT + bs * HD_;
  const float* sr = sinT + bs * HD_;
  size_t obase = (((size_t)b * H_ + h) * S_ + s) * HD_;
  for (int j = 0; j < 64; j++) {
    float cn0 = cr[j] * cd - sr[j] * sd;
    float sn0 = sr[j] * cd + cr[j] * sd;
    float cn1 = cr[j + 64] * cd - sr[j + 64] * sd;
    float sn1 = sr[j + 64] * cd + cr[j + 64] * sd;
    float q0 = qr[j], q1 = qr[j + 64];
    float k0 = kr[j], k1 = kr[j + 64];
    Qr[obase + j]      = (__bf16)(q0 * cn0 - q1 * sn0);
    Qr[obase + j + 64] = (__bf16)(q1 * cn1 + q0 * sn1);
    Kr[obase + j]      = (__bf16)(k0 * cn0 - k1 * sn0);
    Kr[obase + j + 64] = (__bf16)(k1 * cn1 + k0 * sn1);
  }
}

// ---------------------------------------------------------------------------
// V f32 [B,S,H*HD] -> Vt bf16 [B,H,HD,S]  (so P@V B-frags are contiguous)
// ---------------------------------------------------------------------------
__global__ __launch_bounds__(256) void v_transpose(const float* __restrict__ vg,
                                                   __bf16* __restrict__ Vt) {
  int i = blockIdx.x * blockDim.x + threadIdx.x;   // [0, B*H*S)
  if (i >= B_ * H_ * S_) return;
  int b = i / (H_ * S_);
  int h = (i / S_) % H_;
  int s = i % S_;
  const float* vr = vg + ((size_t)b * S_ + s) * (H_ * HD_) + h * HD_;
  __bf16* ob = Vt + (((size_t)b * H_ + h) * HD_) * S_ + s;
  for (int hd = 0; hd < HD_; hd++) ob[(size_t)hd * S_] = (__bf16)vr[hd];
}

// ---------------------------------------------------------------------------
// Block-sparse attention for one (b,h,qblock) per workgroup (8 waves).
// TDM gathers the 6 selected K blocks and V blocks into LDS (1x global
// traffic instead of 4x), then:
//   Phase 1: scores 64x384 via WMMA (Q from global, K from LDS) -> LDS f32
//   Softmax with causal/validity/attention-mask bias in LDS
//   Phase 2: O = P @ V via WMMA (P from LDS w/ f32->bf16 cvt, V from LDS)
// ---------------------------------------------------------------------------
__global__ __launch_bounds__(256) void attn_bs_kernel(const __bf16* __restrict__ Qr,
                                                      const __bf16* __restrict__ Kr,
                                                      const __bf16* __restrict__ Vt,
                                                      const float* __restrict__ amask,
                                                      __bf16* __restrict__ attnB) {
  extern __shared__ char smraw[];
  float* smS = (float*)(smraw + LDS_S_OFF);
  const int wg = blockIdx.x;              // b*H*NB + h*NB + qb
  const int qb = wg % NB_;
  const int h  = (wg / NB_) % H_;
  const int b  = wg / (NB_ * H_);
  const int tid = threadIdx.x, wave = tid >> 5, lane = tid & 31;
  const int lm = lane & 15, g = lane >> 4;

  // --- block selection (matches reference: 2 global strided + 4 local) ---
  int selc[NS_]; int valid[NS_];
  for (int t = 0; t < 2; t++) {
    int gj = qb / 4 + t - 1;
    int gb = gj * 4;
    valid[t] = (gj >= 0) && (gb < qb - 3);
    selc[t] = gb < 0 ? 0 : (gb > NB_ - 1 ? NB_ - 1 : gb);
  }
  for (int t = 0; t < 4; t++) {
    int lb = qb + t - 3;
    valid[2 + t] = (lb >= 0);
    selc[2 + t] = lb < 0 ? 0 : lb;
  }

  const __bf16* Qb = Qr + (((size_t)b * H_ + h) * S_) * HD_;
  const __bf16* Kb = Kr + (((size_t)b * H_ + h) * S_) * HD_;
  const __bf16* Vb = Vt + (((size_t)b * H_ + h) * HD_) * S_;

  // --- TDM gather: selected K blocks [64,128] and V blocks [128,64] -> LDS ---
  if (wave == 0) {
    for (int sb = 0; sb < NS_; sb++) {
      // K block: rows = 64 keys, dim0 (contiguous) = HD_=128 elems
      tdm_load_2d((unsigned)(LDS_K_OFF + sb * 16384),
                  Kb + (size_t)selc[sb] * BS_ * HD_,
                  HD_, S_, HD_, HD_, BS_);
      // V block: rows = 128 hd, dim0 (contiguous) = 64 keys (Vt is [hd][s])
      tdm_load_2d((unsigned)(LDS_V_OFF + sb * 16384),
                  Vb + (size_t)selc[sb] * BS_,
                  S_, HD_, S_, BS_, HD_);
    }
    __builtin_amdgcn_s_wait_tensorcnt(0);
  }
  __syncthreads();

  // --- phase 1: scores. wave w covers key-groups [3w, 3w+3) of 24 ---
  for (int kg = wave * 3; kg < wave * 3 + 3; kg++) {
    int sb = kg >> 2;                                // key block slot 0..5
    int krow = (kg & 3) * 16 + lm;                   // key row within block
    for (int qg = 0; qg < 4; qg++) {
      v8f acc = (v8f)0.f;
      int qrow = qb * BS_ + qg * 16 + lm;
#pragma unroll
      for (int ks = 0; ks < 4; ks++) {               // HD_/32
        FragBF af, bfK;
        const u32x4* pa = (const u32x4*)(Qb + (size_t)qrow * HD_ + ks * 32 + g * 8);
        af.u[0] = pa[0]; af.u[1] = pa[2];
        const u32x4* pb = (const u32x4*)(smraw + LDS_K_OFF + sb * 16384 +
                                         krow * 256 + (ks * 32 + g * 8) * 2);
        bfK.u[0] = pb[0]; bfK.u[1] = pb[2];
        acc = __builtin_amdgcn_wmma_f32_16x16x32_bf16(false, af.v, false, bfK.v,
                                                      (short)0, acc, false, false);
      }
      for (int v = 0; v < 8; v++)
        smS[(qg * 16 + v + 8 * g) * SROW_ + kg * 16 + lm] = acc[v] * SCALE_;
    }
  }
  __syncthreads();

  // --- softmax with bias; one row per thread (tid < 64) ---
  if (tid < 64) {
    int row = tid, qpos = qb * BS_ + row;
    float* srow = smS + (size_t)row * SROW_;
    float mx = -3.4e38f;
    for (int c = 0; c < NSK_; c++) {
      int sb = c >> 6;
      int kpos = selc[sb] * BS_ + (c & 63);
      float bias = (valid[sb] && kpos <= qpos) ? 0.f : NEG_;
      bias += (1.f - amask[(size_t)b * S_ + kpos]) * NEG_;
      float sc = srow[c] + bias;
      srow[c] = sc;
      mx = fmaxf(mx, sc);
    }
    float sum = 0.f;
    for (int c = 0; c < NSK_; c++) { float e = expf(srow[c] - mx); srow[c] = e; sum += e; }
    float inv = 1.f / sum;
    for (int c = 0; c < NSK_; c++) srow[c] *= inv;
  }
  __syncthreads();

  // --- phase 2: O = P @ V. wave w owns hd group w (16 hd cols) ---
  const int hd0 = wave * 16;
  for (int qg = 0; qg < 4; qg++) {
    v8f acc = (v8f)0.f;
    int qrow_l = qg * 16 + lm;
#pragma unroll
    for (int ks = 0; ks < 12; ks++) {                // NSK_/32
      FragBF af, bfV;
      const float* p0 = smS + (size_t)qrow_l * SROW_ + ks * 32 + g * 8;
      const float* p1 = p0 + 16;
#pragma unroll
      for (int j = 0; j < 8; j++) af.v[j] = (__bf16)p0[j];
#pragma unroll
      for (int j = 0; j < 8; j++) af.v[8 + j] = (__bf16)p1[j];
      int kjb = ks * 32 + g * 8;                     // gathered key base (within one block)
      const u32x4* pb = (const u32x4*)(smraw + LDS_V_OFF + (kjb >> 6) * 16384 +
                                       (hd0 + lm) * 128 + (kjb & 63) * 2);
      bfV.u[0] = pb[0]; bfV.u[1] = pb[2];
      acc = __builtin_amdgcn_wmma_f32_16x16x32_bf16(false, af.v, false, bfV.v,
                                                    (short)0, acc, false, false);
    }
    for (int v = 0; v < 8; v++) {
      int row = qb * BS_ + qg * 16 + v + 8 * g;
      attnB[((size_t)b * S_ + row) * (H_ * HD_) + h * HD_ + hd0 + lm] = (__bf16)acc[v];
    }
  }
}

// ---------------------------------------------------------------------------
extern "C" void kernel_launch(void* const* d_in, const int* in_sizes, int n_in,
                              void* d_out, int out_size, void* d_ws, size_t ws_size,
                              hipStream_t stream) {
  (void)in_sizes; (void)n_in; (void)out_size; (void)ws_size;
  const float* hidden = (const float*)d_in[0];
  const float* amask  = (const float*)d_in[1];
  const float* cosT   = (const float*)d_in[2];
  const float* sinT   = (const float*)d_in[3];
  const float* Wq     = (const float*)d_in[4];
  const float* Wk     = (const float*)d_in[5];
  const float* Wv     = (const float*)d_in[6];
  const float* Wo     = (const float*)d_in[7];
  const float* Wph    = (const float*)d_in[8];
  float* out = (float*)d_out;

  const size_t M = (size_t)B_ * S_;       // 8192
  char* ws = (char*)d_ws;
  size_t off = 0;
  auto carve = [&](size_t bytes) -> char* {
    char* p = ws + off; off += (bytes + 255) & ~(size_t)255; return p;
  };
  __bf16* hidB  = (__bf16*)carve(M * D_ * 2);
  __bf16* WqT   = (__bf16*)carve((size_t)D_ * D_ * 2);
  __bf16* WkT   = (__bf16*)carve((size_t)D_ * D_ * 2);
  __bf16* WvT   = (__bf16*)carve((size_t)D_ * D_ * 2);
  __bf16* WoT   = (__bf16*)carve((size_t)D_ * D_ * 2);
  float*  qgm   = (float*)carve(M * D_ * 4);
  float*  kgm   = (float*)carve(M * D_ * 4);
  float*  vgm   = (float*)carve(M * D_ * 4);
  float*  phaseA= (float*)carve(M * H_ * 4);
  __bf16* Qr    = (__bf16*)carve(M * D_ * 2);
  __bf16* Kr    = (__bf16*)carve(M * D_ * 2);
  __bf16* Vtw   = (__bf16*)carve(M * D_ * 2);
  __bf16* attnB = (__bf16*)carve(M * D_ * 2);

  // 1) hidden -> bf16
  {
    int n = (int)(M * D_);
    cvt_f32_bf16<<<n / 256, 256, 0, stream>>>(hidden, hidB, n);
  }
  // 2) weight transposes -> bf16 [N,K]
  {
    int n = D_ * D_;
    transpose_w_bf16<<<n / 256, 256, 0, stream>>>(Wq, WqT, D_, D_);
    transpose_w_bf16<<<n / 256, 256, 0, stream>>>(Wk, WkT, D_, D_);
    transpose_w_bf16<<<n / 256, 256, 0, stream>>>(Wv, WvT, D_, D_);
    transpose_w_bf16<<<n / 256, 256, 0, stream>>>(Wo, WoT, D_, D_);
  }
  // 3) QKV projections (WMMA, 64x64 wave tiles)
  {
    int blocks = (int)((M >> 6) * (D_ >> 6) / 8);   // 512
    gemm_bf16_wmma<<<blocks, 256, 0, stream>>>(hidB, WqT, qgm, (int)M, D_, D_);
    gemm_bf16_wmma<<<blocks, 256, 0, stream>>>(hidB, WkT, kgm, (int)M, D_, D_);
    gemm_bf16_wmma<<<blocks, 256, 0, stream>>>(hidB, WvT, vgm, (int)M, D_, D_);
  }
  // 4) phase projection
  phase_proj<<<(int)(M * H_ / 256), 256, 0, stream>>>(hidden, Wph, phaseA);
  // 5) RoPE -> Qr/Kr bf16 [B,H,S,HD]
  rope_kernel<<<(int)(B_ * H_ * S_ / 256), 256, 0, stream>>>(qgm, kgm, phaseA, cosT, sinT, Qr, Kr);
  // 6) V -> Vt bf16 [B,H,HD,S]
  v_transpose<<<(int)(B_ * H_ * S_ / 256), 256, 0, stream>>>(vgm, Vtw);
  // 7) block-sparse attention (TDM-staged K/V panels)
  {
    int lds_bytes = LDS_TOTAL;                      // 296960 B (< 320 KB WGP LDS)
    (void)hipFuncSetAttribute((const void*)attn_bs_kernel,
                              hipFuncAttributeMaxDynamicSharedMemorySize, lds_bytes);
    attn_bs_kernel<<<B_ * H_ * NB_, 256, lds_bytes, stream>>>(Qr, Kr, Vtw, amask, attnB);
  }
  // 8) output projection -> d_out (f32)
  {
    int blocks = (int)((M >> 6) * (D_ >> 6) / 8);
    gemm_bf16_wmma<<<blocks, 256, 0, stream>>>(attnB, WoT, out, (int)M, D_, D_);
  }
}
